// HFreqCLayer_86011015069999
// MI455X (gfx1250) — matrix operand order repaired and checked by reading, not compiled, not run on hardware
//
#include <hip/hip_runtime.h>
#include <math.h>

// ---- CDNA5 WMMA vector types (wave32) ----
typedef __attribute__((ext_vector_type(16))) _Float16 v16h;
typedef __attribute__((ext_vector_type(8)))  _Float16 v8h;
typedef __attribute__((ext_vector_type(4)))  _Float16 v4h;
typedef __attribute__((ext_vector_type(2)))  _Float16 v2h;
typedef __attribute__((ext_vector_type(8)))  float    v8f;

union V16 { v16h v; v8h h[2]; };

#define C_LEN  728          // FFT length / channels
#define KP     736          // K padded: 23 * 32
#define NP     768          // N padded: 3 * 256 (= 6 * 128)
#define BM     128
#define BNF    256          // fast-path N block tile
#define BN     128          // fallback N block tile
#define BK     32
#define KSTEPS (KP / BK)    // 23
#define LDP    (BK + 8)     // LDS row stride in halves (bank-conflict pad)

// --- CDNA5 async global->LDS copy (ASYNCcnt path), portable via inline asm ---
// VDST = per-lane LDS byte offset (low 32 bits of a generic LDS pointer are the
// LDS offset per the flat-aperture truncation rule), VADDR = 64-bit global addr.
__device__ __forceinline__ unsigned lds_off(const void* p) {
    return (unsigned)(size_t)p;
}
__device__ __forceinline__ void async_b128(unsigned lds, const void* g) {
    asm volatile("global_load_async_to_lds_b128 %0, %1, off"
                 :: "v"(lds), "v"((unsigned long long)(size_t)g) : "memory");
}
__device__ __forceinline__ void wait_async0() {
    asm volatile("s_wait_asynccnt 0" ::: "memory");
}

// ---------------------------------------------------------------------------
// Kernel 1: h[d] = (1/c) * sum_{k kept} cos(2*pi*k*d/c); kept = frequencies
// whose fftshifted index lies outside [c/2 - c/s, c/2 + c/s).
// ---------------------------------------------------------------------------
__global__ void build_h(const int* __restrict__ scale_p, float* __restrict__ h) {
    const int s  = scale_p[0];
    const int lo = C_LEN / 2 - C_LEN / s;
    const int hi = C_LEN / 2 + C_LEN / s;
    const float w = 6.28318530717958647692f / (float)C_LEN;
    for (int d = threadIdx.x; d < C_LEN; d += blockDim.x) {
        float acc = 0.0f;
        for (int k = 0; k < C_LEN; ++k) {
            const int i = (k + C_LEN / 2) % C_LEN;
            if (i < lo || i >= hi) {
                const int t = (k * d) % C_LEN;           // exact arg reduction
                acc += cosf(w * (float)t);
            }
        }
        h[d] = acc / (float)C_LEN;
    }
}

// ---------------------------------------------------------------------------
// Kernel 2: circulant B in f16, n-major [NP][KP]; zero-padded.
// ---------------------------------------------------------------------------
__global__ void build_B(const float* __restrict__ h, _Float16* __restrict__ Bm) {
    const int n = blockIdx.x;
    for (int k = threadIdx.x; k < KP; k += blockDim.x) {
        float v = 0.0f;
        if (n < C_LEN && k < C_LEN) {
            int d = n - k; if (d < 0) d += C_LEN;
            v = h[d];
        }
        Bm[(size_t)n * KP + k] = (_Float16)v;
    }
}

// ---------------------------------------------------------------------------
// Kernel 3 (fast-path prep): X f32 (rows x 728) -> Xh f16 (rows x KP), padded.
// ---------------------------------------------------------------------------
__global__ void convert_X(const float* __restrict__ X, _Float16* __restrict__ Xh,
                          int rows) {
    const size_t idx = ((size_t)blockIdx.x * blockDim.x + threadIdx.x) * 4;
    const size_t total = (size_t)rows * KP;
    if (idx >= total) return;
    const int row = (int)(idx / KP);
    const int k   = (int)(idx % KP);
    v4h o;
    if (k + 3 < C_LEN) {
        const float4 f = *(const float4*)(X + (size_t)row * C_LEN + k);
        o[0] = (_Float16)f.x; o[1] = (_Float16)f.y;
        o[2] = (_Float16)f.z; o[3] = (_Float16)f.w;
    } else {
        #pragma unroll
        for (int j = 0; j < 4; ++j) {
            const int kk = k + j;
            o[j] = (_Float16)((kk < C_LEN) ? X[(size_t)row * C_LEN + kk] : 0.0f);
        }
    }
    *(v4h*)(Xh + idx) = o;
}

// ---------------------------------------------------------------------------
// Fast GEMM: Y = relu(Xh * Bm^T), async-DMA double-buffered 128x256 tiles.
// 8 waves in 2(M) x 4(N) grid; each wave owns 64x64 of C (4x4 WMMA tiles):
// per K-step per wave: 16 v_wmma vs 16 ds_load_b128 (1:1), 6 async DMAs/thread.
// ---------------------------------------------------------------------------
__global__ __launch_bounds__(256) void hifreq_gemm_f16(
        const _Float16* __restrict__ Xh,
        const _Float16* __restrict__ Bm,
        float* __restrict__ Y) {
    __shared__ _Float16 As[2][BM][LDP];
    __shared__ _Float16 Bs[2][BNF][LDP];

    const int tid   = threadIdx.x;
    const int lane  = tid & 31;
    const int wave  = tid >> 5;
    const int waveM = wave >> 2;                 // 0..1 -> +64 rows
    const int waveN = wave & 3;                  // 0..3 -> +64 cols
    const int m0    = blockIdx.x * BM;
    const int n0    = blockIdx.y * BNF;

    // 16B DMA chunks: row stride 4 chunks/row; A = 512 chunks, B = 1024 chunks.
    const int rbase = tid >> 2;                  // 0..63
    const int cb    = (tid & 3) << 3;            // half offset 0,8,16,24

    v8f acc[4][4] = {};

    // Fragment addressing per CDNA5 ISA VGPR layouts (CDNA5 ISA):
    const int mA  = lane & 15;
    const int kbA = (lane < 16) ? 0 : 8;
    const int nB  = lane & 15;
    const int kbB = (lane < 16) ? 0 : 8;

    // prologue: DMA tile 0 into buffer 0
    #pragma unroll
    for (int i = 0; i < 2; ++i)
        async_b128(lds_off(&As[0][rbase + 64 * i][cb]),
                   Xh + (size_t)(m0 + rbase + 64 * i) * KP + cb);
    #pragma unroll
    for (int i = 0; i < 4; ++i)
        async_b128(lds_off(&Bs[0][rbase + 64 * i][cb]),
                   Bm + (size_t)(n0 + rbase + 64 * i) * KP + cb);

    for (int ks = 0; ks < KSTEPS; ++ks) {
        wait_async0();                       // this wave's DMA for tile ks done
        __syncthreads();                     // all waves' DMA done; prev reads done
        if (ks + 1 < KSTEPS) {               // overlap next tile DMA with compute
            const int k0 = (ks + 1) * BK;
            const int nb = (ks + 1) & 1;
            #pragma unroll
            for (int i = 0; i < 2; ++i)
                async_b128(lds_off(&As[nb][rbase + 64 * i][cb]),
                           Xh + (size_t)(m0 + rbase + 64 * i) * KP + k0 + cb);
            #pragma unroll
            for (int i = 0; i < 4; ++i)
                async_b128(lds_off(&Bs[nb][rbase + 64 * i][cb]),
                           Bm + (size_t)(n0 + rbase + 64 * i) * KP + k0 + cb);
        }
        const int b = ks & 1;

        V16 bfrag[4];
        #pragma unroll
        for (int nt = 0; nt < 4; ++nt) {
            const int nn = waveN * 64 + nt * 16 + nB;
            bfrag[nt].h[0] = *(const v8h*)(&Bs[b][nn][kbB]);
            bfrag[nt].h[1] = *(const v8h*)(&Bs[b][nn][kbB + 16]);
        }
        #pragma unroll
        for (int mt = 0; mt < 4; ++mt) {
            const int rr = waveM * 64 + mt * 16 + mA;
            V16 a;
            a.h[0] = *(const v8h*)(&As[b][rr][kbA]);
            a.h[1] = *(const v8h*)(&As[b][rr][kbA + 16]);
            #pragma unroll
            for (int nt = 0; nt < 4; ++nt) {
                acc[mt][nt] = __builtin_amdgcn_wmma_f32_16x16x32_f16(
                    false, a.v, false, bfrag[nt].v,
                    (short)0, acc[mt][nt], false, false);
            }
        }
    }

    const int nOfC = lane & 15;
    const int mAdd = (lane < 16) ? 0 : 8;
    #pragma unroll
    for (int mt = 0; mt < 4; ++mt) {
        #pragma unroll
        for (int nt = 0; nt < 4; ++nt) {
            const int gn = n0 + waveN * 64 + nt * 16 + nOfC;
            if (gn < C_LEN) {
                #pragma unroll
                for (int j = 0; j < 8; ++j) {
                    const int gm = m0 + waveM * 64 + mt * 16 + j + mAdd;
                    const float v = acc[mt][nt][j];
                    Y[(size_t)gm * C_LEN + gn] = v > 0.0f ? v : 0.0f;
                }
            }
        }
    }
}

// ---------------------------------------------------------------------------
// Fallback GEMM (small ws): converts f32 A tiles in-loop, 128x128 tiles.
// ---------------------------------------------------------------------------
__global__ __launch_bounds__(256) void hifreq_gemm_f32(
        const float* __restrict__ X,
        const _Float16* __restrict__ Bm,
        float* __restrict__ Y) {
    __shared__ _Float16 As[BM][LDP];
    __shared__ _Float16 Bs[BN][LDP];

    const int tid   = threadIdx.x;
    const int lane  = tid & 31;
    const int wave  = tid >> 5;
    const int waveM = wave >> 2;
    const int waveN = wave & 3;
    const int m0    = blockIdx.x * BM;
    const int n0    = blockIdx.y * BN;

    v8f acc[4][2] = {};
    const int mA  = lane & 15;
    const int kbA = (lane < 16) ? 0 : 8;
    const int nB  = lane & 15;
    const int kbB = (lane < 16) ? 0 : 8;

    for (int ks = 0; ks < KSTEPS; ++ks) {
        const int k0 = ks * BK;
        #pragma unroll
        for (int i = 0; i < 8; ++i) {
            const int lin = tid + i * 256;
            const int row = lin >> 4;
            const int c2  = (lin & 15) << 1;
            const int gk  = k0 + c2;
            float2 v = make_float2(0.0f, 0.0f);
            if (gk < C_LEN)
                v = *(const float2*)(X + (size_t)(m0 + row) * C_LEN + gk);
            v2h p; p[0] = (_Float16)v.x; p[1] = (_Float16)v.y;
            *(v2h*)(&As[row][c2]) = p;
        }
        #pragma unroll
        for (int i = 0; i < 2; ++i) {
            const int lin  = tid + i * 256;
            const int nrow = lin >> 2;
            const int kc   = (lin & 3) << 3;
            const v8h v = *(const v8h*)(Bm + (size_t)(n0 + nrow) * KP + k0 + kc);
            *(v8h*)(&Bs[nrow][kc]) = v;
        }
        __syncthreads();

        V16 bfrag[2];
        #pragma unroll
        for (int nt = 0; nt < 2; ++nt) {
            const int nn = waveN * 32 + nt * 16 + nB;
            bfrag[nt].h[0] = *(const v8h*)(&Bs[nn][kbB]);
            bfrag[nt].h[1] = *(const v8h*)(&Bs[nn][kbB + 16]);
        }
        #pragma unroll
        for (int mt = 0; mt < 4; ++mt) {
            const int rr = waveM * 64 + mt * 16 + mA;
            V16 a;
            a.h[0] = *(const v8h*)(&As[rr][kbA]);
            a.h[1] = *(const v8h*)(&As[rr][kbA + 16]);
            #pragma unroll
            for (int nt = 0; nt < 2; ++nt) {
                acc[mt][nt] = __builtin_amdgcn_wmma_f32_16x16x32_f16(
                    false, a.v, false, bfrag[nt].v,
                    (short)0, acc[mt][nt], false, false);
            }
        }
        __syncthreads();
    }

    const int nOfC = lane & 15;
    const int mAdd = (lane < 16) ? 0 : 8;
    #pragma unroll
    for (int mt = 0; mt < 4; ++mt) {
        #pragma unroll
        for (int nt = 0; nt < 2; ++nt) {
            const int gn = n0 + waveN * 32 + nt * 16 + nOfC;
            if (gn < C_LEN) {
                #pragma unroll
                for (int j = 0; j < 8; ++j) {
                    const int gm = m0 + waveM * 64 + mt * 16 + j + mAdd;
                    const float v = acc[mt][nt][j];
                    Y[(size_t)gm * C_LEN + gn] = v > 0.0f ? v : 0.0f;
                }
            }
        }
    }
}

// ---------------------------------------------------------------------------
extern "C" void kernel_launch(void* const* d_in, const int* in_sizes, int n_in,
                              void* d_out, int out_size, void* d_ws, size_t ws_size,
                              hipStream_t stream) {
    const float* x     = (const float*)d_in[0];
    const int*   scale = (const int*)d_in[1];
    float*       y     = (float*)d_out;

    const int rows = in_sizes[0] / C_LEN;                // 46208 = 361 * 128

    // ws layout: h | circulant f16 | (fast path) f16 copy of x
    const size_t offB = 4096;
    const size_t offX = offB + (size_t)NP * KP * sizeof(_Float16);
    const size_t need = offX + (size_t)rows * KP * sizeof(_Float16);

    float*    h  = (float*)d_ws;
    _Float16* Bm = (_Float16*)((char*)d_ws + offB);
    _Float16* Xh = (_Float16*)((char*)d_ws + offX);

    build_h<<<1, 256, 0, stream>>>(scale, h);
    build_B<<<NP, 256, 0, stream>>>(h, Bm);

    if (ws_size >= need) {
        const size_t tot4 = ((size_t)rows * KP) / 4;
        convert_X<<<(unsigned)((tot4 + 255) / 256), 256, 0, stream>>>(x, Xh, rows);
        dim3 grid(rows / BM, NP / BNF);                  // 361 x 3
        hifreq_gemm_f16<<<grid, 256, 0, stream>>>(Xh, Bm, y);
    } else {
        dim3 grid(rows / BM, (C_LEN + BN - 1) / BN);     // 361 x 6
        hifreq_gemm_f32<<<grid, 256, 0, stream>>>(x, Bm, y);
    }
}
